// VoxelFeatureExtractor_5549097746509
// MI455X (gfx1250) — compile-verified, matrix operand order checked by reading, not run on hardware
//
#include <hip/hip_runtime.h>

// ---------------------------------------------------------------------------
// VoxelFeatureExtractor for MI455X (gfx1250, wave32).
//   - GEMMs: v_wmma_f32_16x16x32_bf16, 8 waves/block = 64 rows, 4(M)x2(N) waves
//   - Weights pre-converted once to bf16 in the exact [chunk][col][k] LDS image,
//     then staged per K-chunk with global_load_async_to_lds_b128 (ASYNCcnt),
//     double-buffered so the next chunk's DMA overlaps current WMMAs.
//   - BN statistics fused into GEMM epilogue (shfl + per-column atomics).
//   - Segment max pooling with order-preserving u32 encoding + atomicMax.
// ---------------------------------------------------------------------------

typedef __attribute__((ext_vector_type(16))) __bf16 v16bf;
typedef __attribute__((ext_vector_type(8)))  __bf16 v8bf;
typedef __attribute__((ext_vector_type(8)))  float  v8f;

union FragBF { v16bf v; v8bf h[2]; };

static constexpr int   Nn   = 262144;
static constexpr float EPSf = 1e-5f;

// ---- bf16 <-> f32 (manual, RNE) -------------------------------------------
__device__ __forceinline__ float bf2f(__bf16 h) {
  unsigned short u = __builtin_bit_cast(unsigned short, h);
  return __uint_as_float(((unsigned)u) << 16);
}
__device__ __forceinline__ __bf16 f2bf(float f) {
  unsigned u = __float_as_uint(f);
  unsigned short r;
  if (((u >> 23) & 0xffu) == 0xffu) r = (unsigned short)(u >> 16);      // inf/nan
  else r = (unsigned short)((u + 0x7fffu + ((u >> 16) & 1u)) >> 16);    // RNE
  return __builtin_bit_cast(__bf16, r);
}

// ---- order-preserving float<->u32 encoding for atomic max -----------------
__device__ __forceinline__ unsigned encF(float f) {
  unsigned u = __float_as_uint(f);
  return (u & 0x80000000u) ? ~u : (u | 0x80000000u);
}
__device__ __forceinline__ float decF(unsigned e) {
  unsigned u = (e & 0x80000000u) ? (e & 0x7fffffffu) : ~e;
  return __uint_as_float(u);
}

// ---------------------------------------------------------------------------
// init: zero voxel counts + BN stats accumulators (+ mask count)
// ---------------------------------------------------------------------------
__global__ __launch_bounds__(256) void vfe_init_misc(int* __restrict__ cnt,
                                                     float* __restrict__ stats) {
  int gid = blockIdx.x * 256 + threadIdx.x;
  if (gid < Nn)  cnt[gid]  = 0;
  if (gid < 915) stats[gid] = 0.0f;
}

__global__ __launch_bounds__(256) void vfe_init_pool(unsigned* __restrict__ pool) {
  long long gid = (long long)blockIdx.x * 256 + threadIdx.x;
  pool[gid] = 0u;   // decodes to "very small"; only rows < U are ever read
}

// ---------------------------------------------------------------------------
// weight prep: f32 [kReal][F] -> bf16 chunked LDS image [kc][col][k], k<32,
// zero-padded for rows >= kReal. 16-byte aligned, DMA-ready.
// ---------------------------------------------------------------------------
__global__ __launch_bounds__(256) void vfe_wprep(const float* __restrict__ W,
                                                 __bf16* __restrict__ dst,
                                                 int kReal, int F, int KC) {
  int id = blockIdx.x * 256 + threadIdx.x;
  int total = KC * F * 32;
  if (id >= total) return;
  int k   = id & 31;
  int col = (id >> 5) % F;
  int kc  = (id >> 5) / F;
  int wr  = kc * 32 + k;
  float v = (wr < kReal) ? W[(long long)wr * F + col] : 0.0f;
  dst[((long long)kc * F + col) * 32 + k] = f2bf(v);
}

// ---------------------------------------------------------------------------
// grouping: key, per-voxel count, arrival rank (mask = grp < 64)
// ---------------------------------------------------------------------------
__global__ __launch_bounds__(256) void vfe_group(const int* __restrict__ pt_ind,
                                                 int* __restrict__ cnt,
                                                 int* __restrict__ grp,
                                                 int* __restrict__ key) {
  int i = blockIdx.x * 256 + threadIdx.x;
  if (i >= Nn) return;
  int b = pt_ind[i * 4 + 0], x = pt_ind[i * 4 + 1];
  int y = pt_ind[i * 4 + 2], z = pt_ind[i * 4 + 3];
  int k = ((b * 64 + x) * 64 + y) * 16 + z;
  key[i] = k;
  grp[i] = atomicAdd(&cnt[k], 1);
}

// ---- scan over "present" flags: rank[key] = position among sorted uniques --
__global__ __launch_bounds__(256) void vfe_scanA(const int* __restrict__ cnt,
                                                 int* __restrict__ rank,
                                                 int* __restrict__ bsum) {
  __shared__ int sd[256];
  int tid = threadIdx.x;
  int base = blockIdx.x * 1024 + tid * 4;
  int p[4], ts = 0;
#pragma unroll
  for (int j = 0; j < 4; ++j) { p[j] = (cnt[base + j] > 0) ? 1 : 0; ts += p[j]; }
  sd[tid] = ts; __syncthreads();
  for (int off = 1; off < 256; off <<= 1) {
    int t = (tid >= off) ? sd[tid - off] : 0;
    __syncthreads(); sd[tid] += t; __syncthreads();
  }
  int excl = sd[tid] - ts, run = excl;
#pragma unroll
  for (int j = 0; j < 4; ++j) { rank[base + j] = run; run += p[j]; }
  if (tid == 255) bsum[blockIdx.x] = sd[255];
}

__global__ __launch_bounds__(256) void vfe_scanB(const int* __restrict__ bsum,
                                                 int* __restrict__ boff,
                                                 int* __restrict__ dU) {
  __shared__ int sd[256];
  int tid = threadIdx.x;
  int v = bsum[tid];
  sd[tid] = v; __syncthreads();
  for (int off = 1; off < 256; off <<= 1) {
    int t = (tid >= off) ? sd[tid - off] : 0;
    __syncthreads(); sd[tid] += t; __syncthreads();
  }
  boff[tid] = sd[tid] - v;
  if (tid == 255) dU[0] = sd[255];
}

__global__ __launch_bounds__(256) void vfe_scanC(int* __restrict__ rank,
                                                 const int* __restrict__ boff) {
  int base = blockIdx.x * 1024 + threadIdx.x * 4;
  int o = boff[blockIdx.x];
#pragma unroll
  for (int j = 0; j < 4; ++j) rank[base + j] += o;
}

// ---- coords output (int64) -------------------------------------------------
__global__ __launch_bounds__(256) void vfe_coords(const int* __restrict__ cnt,
                                                  const int* __restrict__ rank,
                                                  long long* __restrict__ outC) {
  int k = blockIdx.x * 256 + threadIdx.x;
  if (k >= Nn || cnt[k] <= 0) return;
  long long j = rank[k];
  int z = k & 15, t = k >> 4;
  int y = t & 63; t >>= 6;
  int x = t & 63; int b = t >> 6;
  outC[j * 4 + 0] = b; outC[j * 4 + 1] = x;
  outC[j * 4 + 2] = y; outC[j * 4 + 3] = z;
}

__global__ __launch_bounds__(256) void vfe_pad(const int* __restrict__ dU,
                                               long long* __restrict__ outC) {
  int r = blockIdx.x * 256 + threadIdx.x;
  if (r >= Nn || r < dU[0]) return;
#pragma unroll
  for (int c = 0; c < 4; ++c) outC[(long long)r * 4 + c] = -1LL;
}

// ---------------------------------------------------------------------------
// bn0 stats (masked sum/sumsq over 9 dims + mask count)
// ---------------------------------------------------------------------------
__device__ __forceinline__ void blkAdd(float v, float* red, int tid, float* dst) {
  red[tid] = v; __syncthreads();
  for (int o = 128; o > 0; o >>= 1) { if (tid < o) red[tid] += red[tid + o]; __syncthreads(); }
  if (tid == 0) atomicAdd(dst, red[0]);
  __syncthreads();
}

__global__ __launch_bounds__(256) void vfe_bn0stats(const float* __restrict__ pf,
                                                    const int* __restrict__ grp,
                                                    float* __restrict__ sum9,
                                                    float* __restrict__ sq9,
                                                    float* __restrict__ nmask) {
  __shared__ float red[256];
  int tid = threadIdx.x;
  long long i = (long long)blockIdx.x * 256 + tid;
  float m = (grp[i] < 64) ? 1.0f : 0.0f;
  float s[9], q[9];
#pragma unroll
  for (int c = 0; c < 9; ++c) {
    float x = pf[i * 9 + c];
    s[c] = x * m; q[c] = x * x * m;
  }
#pragma unroll
  for (int c = 0; c < 9; ++c) { blkAdd(s[c], red, tid, &sum9[c]); blkAdd(q[c], red, tid, &sq9[c]); }
  blkAdd(m, red, tid, nmask);
}

__global__ void vfe_bnfin(const float* __restrict__ sum, const float* __restrict__ sq,
                          const float* __restrict__ nm, const float* __restrict__ g,
                          const float* __restrict__ b, float* __restrict__ sc,
                          float* __restrict__ sh, int F) {
  int c = threadIdx.x;
  if (c >= F) return;
  float n = nm[0];
  float mean = sum[c] / n;
  float var  = sq[c] / n - mean * mean;
  float inv  = rsqrtf(var + EPSf);
  sc[c] = inv * g[c];
  sh[c] = b[c] - mean * inv * g[c];
}

// ---------------------------------------------------------------------------
// Fused GEMM pass:  Y = transform_in(X) @ W + bias  (+ masked stats epilogue)
//   MODE 0: f32 input, BN scale/shift (layer 1)
//   MODE 1: bf16 input, BN scale/shift + relu (layers 2..4)
//   MODE 2: encoded-u32 pooled input, 0 for rows >= U (final layer)
// Weights arrive pre-converted (bf16, [chunk][col][k]); each K-chunk is DMA'd
// into double-buffered LDS with global_load_async_to_lds_b128 so the next
// chunk's copy overlaps the current chunk's WMMAs.
// ---------------------------------------------------------------------------
template <int FIN, int FOUT, int MODE, bool STATS, bool ORELU, bool OBF16>
__global__ __launch_bounds__(256) void vfe_gemm(
    const void* __restrict__ Xv, const float* __restrict__ inScale,
    const float* __restrict__ inShift, const __bf16* __restrict__ Wp,
    const float* __restrict__ bias, void* __restrict__ Yv,
    float* __restrict__ oSum, float* __restrict__ oSq,
    const int* __restrict__ grp, const int* __restrict__ dU, int kReal) {
  constexpr int KC = FIN / 32;
  constexpr int NT = FOUT / 32;
  __shared__ __align__(16) __bf16 Xn[64 * 32];
  __shared__ __align__(16) __bf16 Wt[2][FOUT * 32];

  const int tid  = threadIdx.x;
  const int lane = tid & 31;
  const int wave = tid >> 5;
  const int msub = wave >> 1;
  const int nhalf = wave & 1;
  const long long rowBase = (long long)blockIdx.x * 64;

  int Uv = 0;
  if constexpr (MODE == 2) Uv = dU[0];

  v8f acc[NT] = {};

  const int srow = tid >> 2;
  const int scg  = (tid & 3) << 3;
  constexpr int NVEC = FOUT * 4;   // 16-byte vectors per weight K-chunk

  // async DMA one K-chunk of pre-layouted weights into LDS buffer `buf`
  auto issueW = [&](int kc, int buf) {
    unsigned long long src = (unsigned long long)Wp + (unsigned long long)kc * FOUT * 64;
    for (int e = tid; e < NVEC; e += 256) {
      unsigned ldsoff = (unsigned)(size_t)&Wt[buf][e * 8];
      unsigned long long g = src + (unsigned)(e * 16);
      asm volatile("global_load_async_to_lds_b128 %0, %1, off"
                   :: "v"(ldsoff), "v"(g) : "memory");
    }
  };

  issueW(0, 0);
  int cur = 0;

  for (int kc = 0; kc < KC; ++kc) {
    const int k0 = kc * 32;
    __syncthreads();                                  // prev readers of Xn/Wt done
    asm volatile("s_wait_asynccnt 0" ::: "memory");   // own copies of Wt[cur] landed
    // ---- stage normalized input tile (64 x 32, bf16) ----
    {
      const long long gr = rowBase + srow;
#pragma unroll
      for (int j = 0; j < 8; ++j) {
        const int c = k0 + scg + j;
        float val = 0.0f;
        if constexpr (MODE == 0) {
          if (c < kReal) {
            float x = ((const float*)Xv)[gr * kReal + c];
            val = fmaf(x, inScale[c], inShift[c]);
          }
        } else if constexpr (MODE == 1) {
          float x = bf2f(((const __bf16*)Xv)[gr * FIN + c]);
          val = fmaxf(fmaf(x, inScale[c], inShift[c]), 0.0f);
        } else {
          if (gr < (long long)Uv) val = decF(((const unsigned*)Xv)[gr * FIN + c]);
        }
        Xn[srow * 32 + scg + j] = f2bf(val);
      }
    }
    __syncthreads();                                  // Xn + whole Wt[cur] visible
    if (kc + 1 < KC) issueW(kc + 1, cur ^ 1);         // overlap next DMA with WMMAs

    // ---- fragments + WMMA ----
    const int r  = lane & 15;
    const int kb = (lane >> 4) * 8;
    FragBF a;
    a.h[0] = *(const v8bf*)&Xn[(msub * 16 + r) * 32 + kb];
    a.h[1] = *(const v8bf*)&Xn[(msub * 16 + r) * 32 + kb + 16];
#pragma unroll
    for (int t = 0; t < NT; ++t) {
      const int colb = nhalf * (FOUT / 2) + t * 16;
      FragBF bfr;
      bfr.h[0] = *(const v8bf*)&Wt[cur][(colb + r) * 32 + kb];
      bfr.h[1] = *(const v8bf*)&Wt[cur][(colb + r) * 32 + kb + 16];
      acc[t] = __builtin_amdgcn_wmma_f32_16x16x32_bf16(
          false, a.v, false, bfr.v, (short)0, acc[t], false, false);
    }
    cur ^= 1;
  }

  // ---- epilogue: bias (+relu), store, masked stats ----
  const int r = lane & 15;
  const int rowoff = (lane < 16) ? 0 : 8;
  const long long gr0 = rowBase + msub * 16 + rowoff;
#pragma unroll
  for (int t = 0; t < NT; ++t) {
    const int col = nhalf * (FOUT / 2) + t * 16 + r;
    const float bv = bias[col];
    float s = 0.0f, sq = 0.0f;
#pragma unroll
    for (int e = 0; e < 8; ++e) {
      const long long gr = gr0 + e;
      float v = acc[t][e] + bv;
      if constexpr (ORELU) v = fmaxf(v, 0.0f);
      if constexpr (OBF16) ((__bf16*)Yv)[gr * FOUT + col] = f2bf(v);
      else                 ((float*)Yv)[gr * FOUT + col] = v;
      if constexpr (STATS) {
        const float m = (grp[gr] < 64) ? 1.0f : 0.0f;
        s  += v * m;
        sq += v * v * m;
      }
    }
    if constexpr (STATS) {
      s  += __shfl_xor(s, 16, 32);
      sq += __shfl_xor(sq, 16, 32);
      if (lane < 16) { atomicAdd(&oSum[col], s); atomicAdd(&oSq[col], sq); }
    }
  }
}

// ---------------------------------------------------------------------------
// pooling: per-voxel max over masked points (native u32 atomic max)
// ---------------------------------------------------------------------------
__global__ __launch_bounds__(256) void vfe_pool(const __bf16* __restrict__ h4,
                                                const int* __restrict__ key,
                                                const int* __restrict__ grp,
                                                const int* __restrict__ rank,
                                                unsigned* __restrict__ pool) {
  long long i = blockIdx.x;
  int c = threadIdx.x;
  if (grp[i] >= 64) return;
  long long j = rank[key[i]];
  float v = bf2f(h4[i * 256 + c]);
  atomicMax(&pool[j * 256 + c], encF(v));
}

// ---------------------------------------------------------------------------
extern "C" void kernel_launch(void* const* d_in, const int* in_sizes, int n_in,
                              void* d_out, int out_size, void* d_ws, size_t ws_size,
                              hipStream_t stream) {
  (void)in_sizes; (void)n_in; (void)out_size; (void)ws_size;
  const float* pt_fea = (const float*)d_in[0];
  const int*   pt_ind = (const int*)d_in[1];
  const float* bn0g = (const float*)d_in[2],  *bn0b = (const float*)d_in[3];
  const float* w1 = (const float*)d_in[4],    *b1 = (const float*)d_in[5];
  const float* bn1g = (const float*)d_in[6],  *bn1b = (const float*)d_in[7];
  const float* w2 = (const float*)d_in[8],    *b2 = (const float*)d_in[9];
  const float* bn2g = (const float*)d_in[10], *bn2b = (const float*)d_in[11];
  const float* w3 = (const float*)d_in[12],   *b3 = (const float*)d_in[13];
  const float* bn3g = (const float*)d_in[14], *bn3b = (const float*)d_in[15];
  const float* w4 = (const float*)d_in[16],   *b4 = (const float*)d_in[17];
  const float* wc = (const float*)d_in[18],   *bc = (const float*)d_in[19];

  char* ws = (char*)d_ws;
  const size_t SZ_ACT = (size_t)Nn * 256 * 2;        // bf16 N x 256
  void*     bufB = (void*)(ws + 0);                  // h2 / h4
  void*     bufA = (void*)(ws + SZ_ACT);             // h1 / h3 (dead after L4)
  unsigned* pool = (unsigned*)(ws + SZ_ACT);         // overlaps bufA (re-init post-L4)
  size_t o = SZ_ACT + (size_t)Nn * 256 * 4;          // end of pool region
  int* cnt  = (int*)(ws + o); o += (size_t)Nn * 4;
  int* rank = (int*)(ws + o); o += (size_t)Nn * 4;
  int* key  = (int*)(ws + o); o += (size_t)Nn * 4;
  int* grp  = (int*)(ws + o); o += (size_t)Nn * 4;
  int* bsum = (int*)(ws + o); o += 1024;
  int* boff = (int*)(ws + o); o += 1024;
  int* dU   = (int*)(ws + o); o += 256;
  float* stats = (float*)(ws + o); o += 8192;        // 1829 floats used
  o = (o + 255) & ~(size_t)255;
  __bf16* wp1 = (__bf16*)(ws + o); o += 4096;        // 1x64x32
  __bf16* wp2 = (__bf16*)(ws + o); o += 16384;       // 2x128x32
  __bf16* wp3 = (__bf16*)(ws + o); o += 65536;       // 4x256x32
  __bf16* wp4 = (__bf16*)(ws + o); o += 131072;      // 8x256x32
  __bf16* wpc = (__bf16*)(ws + o); o += 65536;       // 8x128x32

  float* nmask = stats + 0;
  float* S0s = stats + 1,   *S0q = stats + 10;       // 9
  float* S1s = stats + 19,  *S1q = stats + 83;       // 64
  float* S2s = stats + 147, *S2q = stats + 275;      // 128
  float* S3s = stats + 403, *S3q = stats + 659;      // 256  (zero region = 915 floats)
  float* sc0 = stats + 915,  *sh0 = stats + 924;
  float* sc1 = stats + 933,  *sh1 = stats + 997;
  float* sc2 = stats + 1061, *sh2 = stats + 1189;
  float* sc3 = stats + 1317, *sh3 = stats + 1573;

  long long* outCoords = (long long*)d_out;
  float* feat = (float*)((char*)d_out + (size_t)Nn * 4 * sizeof(long long));

  const dim3 T(256);

  // weight prep (tiny; reads only d_in)
  vfe_wprep<<<8,   T, 0, stream>>>(w1, wp1, 9,   64,  1);
  vfe_wprep<<<32,  T, 0, stream>>>(w2, wp2, 64,  128, 2);
  vfe_wprep<<<128, T, 0, stream>>>(w3, wp3, 128, 256, 4);
  vfe_wprep<<<256, T, 0, stream>>>(w4, wp4, 256, 256, 8);
  vfe_wprep<<<128, T, 0, stream>>>(wc, wpc, 256, 128, 8);

  vfe_init_misc<<<1024, T, 0, stream>>>(cnt, stats);
  vfe_group<<<1024, T, 0, stream>>>(pt_ind, cnt, grp, key);
  vfe_scanA<<<256, T, 0, stream>>>(cnt, rank, bsum);
  vfe_scanB<<<1, T, 0, stream>>>(bsum, boff, dU);
  vfe_scanC<<<256, T, 0, stream>>>(rank, boff);
  vfe_coords<<<1024, T, 0, stream>>>(cnt, rank, outCoords);
  vfe_pad<<<1024, T, 0, stream>>>(dU, outCoords);

  vfe_bn0stats<<<1024, T, 0, stream>>>(pt_fea, grp, S0s, S0q, nmask);
  vfe_bnfin<<<1, T, 0, stream>>>(S0s, S0q, nmask, bn0g, bn0b, sc0, sh0, 9);

  // L1: bn0(pt_fea) @ w1 + b1 -> h1 (bf16), stats for bn1
  vfe_gemm<32, 64, 0, true, false, true><<<Nn / 64, T, 0, stream>>>(
      pt_fea, sc0, sh0, wp1, b1, bufA, S1s, S1q, grp, dU, 9);
  vfe_bnfin<<<1, T, 0, stream>>>(S1s, S1q, nmask, bn1g, bn1b, sc1, sh1, 64);

  // L2: relu(bn1(h1)) @ w2 + b2 -> h2, stats for bn2
  vfe_gemm<64, 128, 1, true, false, true><<<Nn / 64, T, 0, stream>>>(
      bufA, sc1, sh1, wp2, b2, bufB, S2s, S2q, grp, dU, 64);
  vfe_bnfin<<<1, T, 0, stream>>>(S2s, S2q, nmask, bn2g, bn2b, sc2, sh2, 128);

  // L3: relu(bn2(h2)) @ w3 + b3 -> h3, stats for bn3
  vfe_gemm<128, 256, 1, true, false, true><<<Nn / 64, T, 0, stream>>>(
      bufB, sc2, sh2, wp3, b3, bufA, S3s, S3q, grp, dU, 128);
  vfe_bnfin<<<1, T, 0, stream>>>(S3s, S3q, nmask, bn3g, bn3b, sc3, sh3, 256);

  // L4: relu(bn3(h3)) @ w4 + b4 -> h4 (no stats)
  vfe_gemm<256, 256, 1, false, false, true><<<Nn / 64, T, 0, stream>>>(
      bufA, sc3, sh3, wp4, b4, bufB, nullptr, nullptr, grp, dU, 256);

  // pooling (pool overlaps dead h1/h3 arena; re-zero now, then segment max)
  vfe_init_pool<<<Nn, T, 0, stream>>>(pool);
  vfe_pool<<<Nn, T, 0, stream>>>((const __bf16*)bufB, key, grp, rank, pool);

  // final: relu(pooled @ wc + bc) -> features (f32)
  vfe_gemm<256, 128, 2, false, true, false><<<Nn / 64, T, 0, stream>>>(
      pool, nullptr, nullptr, wpc, bc, feat, nullptr, nullptr, grp, dU, 256);
}
